// ActivePredictiveLayer_60266981097497
// MI455X (gfx1250) — matrix operand order repaired and checked by reading, not compiled, not run on hardware
//
#include <hip/hip_runtime.h>
#include <cstdint>

// ---------------------------------------------------------------------------
// ActivePredictiveLayer for MI455X (gfx1250, wave32, WMMA).
//
// free_energy GEMM + 10 Langevin steps (step 0 elementwise, 9 GEMM), all fused epilogues.
// FP32 GEMMs implemented via split-bf16 (hi+lo) V_WMMA_F32_16X16X32_BF16:
// 3 bf16 WMMAs per tile ~= fp32 accuracy at ~2.7x the native f32-WMMA rate.
// ---------------------------------------------------------------------------

typedef __attribute__((ext_vector_type(16))) __bf16 v16bf;
typedef __attribute__((ext_vector_type(8)))  float  v8f;

#define TOKENS 8192
#define FEATS  4096
#define DT_C      0.1f
#define BASE_TEMP 0.1f

#define BM 128
#define BN 128
#define BK 32
#define LDK 36          // 32 + 4 bf16 pad -> row stride 72B, spreads LDS banks

__device__ __forceinline__ void f32_to_bf16x2(float x, __bf16& hi, __bf16& lo) {
  hi = (__bf16)x;                 // RNE truncation to bf16
  lo = (__bf16)(x - (float)hi);   // residual
}

// Deterministic per-(step, element) standard normal: splitmix64 + Box-Muller.
__device__ __forceinline__ float gauss_noise(uint64_t s) {
  s += 0x9E3779B97F4A7C15ull;
  s = (s ^ (s >> 30)) * 0xBF58476D1CE4E5B9ull;
  s = (s ^ (s >> 27)) * 0x94D049BB133111EBull;
  s = s ^ (s >> 31);
  unsigned int a = (unsigned int)s;
  unsigned int b = (unsigned int)(s >> 32);
  float u1 = ((float)a + 0.5f) * 2.3283064365386963e-10f;  // (0,1)
  float u2 = (float)b * 2.3283064365386963e-10f;
  return sqrtf(-2.0f * __logf(u1)) * __cosf(6.28318530718f * u2);
}

__device__ __forceinline__ float noise_scale_from(const float* sumsq) {
  float fe = *sumsq * (1.0f / ((float)TOKENS * (float)FEATS)); // mean(err^2)
  float temp = BASE_TEMP * (1.0f + fe * 10.0f);
  return sqrtf(2.0f * DT_C * temp);
}

// --- WMMA fragment gathers from LDS (contiguous k order, same for A and B) ------
// A 16x32 bf16: M = lane&15; halves e: K = e + 16*(lane>>4)
__device__ __forceinline__ v16bf load_a_frag(const __bf16* sA, int mBase, int lane) {
  const __bf16* p = sA + (size_t)(mBase + (lane & 15)) * LDK;
  const int koff = (lane >> 4) << 4;   // 0 or 16
  v16bf f;
#pragma unroll
  for (int e = 0; e < 16; ++e) f[e] = p[koff + e];
  return f;
}
// B 32x16 bf16 (LDS holds B transposed: [n][k]): N = lane&15; K = e + 16*(lane>>4)
__device__ __forceinline__ v16bf load_b_frag(const __bf16* sBt, int nBase, int lane) {
  const __bf16* p = sBt + (size_t)(nBase + (lane & 15)) * LDK + ((lane >> 4) << 4);
  v16bf f;
#pragma unroll
  for (int e = 0; e < 16; ++e) f[e] = p[e];
  return f;
}

// MODE 0: free-energy GEMM  (A=x_input, B=W):  atomicAdd sum((x - tanh(A@B))^2)
// MODE 1: Langevin step     (A=x_cur,   B=J):  out = tanh(x - (A@B - x_in)*DT + noise)
template <int MODE>
__global__ __launch_bounds__(256)
void fused_gemm_kernel(const float* __restrict__ A,
                       const float* __restrict__ B,
                       const float* __restrict__ xin,
                       float* __restrict__ out,
                       float* __restrict__ sumsq,
                       int step)
{
  __shared__ __bf16 sAhi[BM * LDK];
  __shared__ __bf16 sAlo[BM * LDK];
  __shared__ __bf16 sBhi[BN * LDK];
  __shared__ __bf16 sBlo[BN * LDK];

  const int t    = threadIdx.x;
  const int lane = t & 31;
  const int wave = t >> 5;           // 8 waves
  const int wm   = wave >> 2;        // 0..1  -> 64 rows each
  const int wn   = wave & 3;         // 0..3  -> 32 cols each
  const int blockM = blockIdx.y * BM;
  const int blockN = blockIdx.x * BN;

  v8f acc[4][2];
#pragma unroll
  for (int i = 0; i < 4; ++i)
#pragma unroll
    for (int j = 0; j < 2; ++j)
#pragma unroll
      for (int e = 0; e < 8; ++e) acc[i][j][e] = 0.0f;

  for (int k0 = 0; k0 < FEATS; k0 += BK) {
    __syncthreads();   // protect LDS from previous iteration's readers
    // ---- stage A tile 128x32 (row-major, split hi/lo) ----
    {
      const int acol = (t & 7) << 2;   // 0..28
      const int arow = t >> 3;         // 0..31
#pragma unroll
      for (int i = 0; i < 4; ++i) {
        const int r = arow + 32 * i;
        const float4 v = *(const float4*)(A + (size_t)(blockM + r) * FEATS + k0 + acol);
        const float vv[4] = {v.x, v.y, v.z, v.w};
#pragma unroll
        for (int j = 0; j < 4; ++j) {
          __bf16 hi, lo; f32_to_bf16x2(vv[j], hi, lo);
          sAhi[(size_t)r * LDK + acol + j] = hi;
          sAlo[(size_t)r * LDK + acol + j] = lo;
        }
      }
    }
    // ---- stage B tile 32x128, transposed into [n][k] (split hi/lo) ----
    {
      const int bcol = (t & 31) << 2;  // 0..124
      const int brow = t >> 5;         // 0..7
#pragma unroll
      for (int i = 0; i < 4; ++i) {
        const int kk = brow + 8 * i;
        const float4 v = *(const float4*)(B + (size_t)(k0 + kk) * FEATS + blockN + bcol);
        const float vv[4] = {v.x, v.y, v.z, v.w};
#pragma unroll
        for (int j = 0; j < 4; ++j) {
          __bf16 hi, lo; f32_to_bf16x2(vv[j], hi, lo);
          sBhi[(size_t)(bcol + j) * LDK + kk] = hi;
          sBlo[(size_t)(bcol + j) * LDK + kk] = lo;
        }
      }
    }
    __syncthreads();

    // ---- WMMA: acc += Ahi*Bhi + Ahi*Blo + Alo*Bhi ----
    v16bf ahi[4], alo[4];
#pragma unroll
    for (int mt = 0; mt < 4; ++mt) {
      const int mBase = wm * 64 + mt * 16;
      ahi[mt] = load_a_frag(sAhi, mBase, lane);
      alo[mt] = load_a_frag(sAlo, mBase, lane);
    }
#pragma unroll
    for (int nt = 0; nt < 2; ++nt) {
      const int nBase = wn * 32 + nt * 16;
      const v16bf bhi = load_b_frag(sBhi, nBase, lane);
      const v16bf blo = load_b_frag(sBlo, nBase, lane);
#pragma unroll
      for (int mt = 0; mt < 4; ++mt) {
        acc[mt][nt] = __builtin_amdgcn_wmma_f32_16x16x32_bf16(
            false, ahi[mt], false, bhi, (short)0, acc[mt][nt], false, false);
        acc[mt][nt] = __builtin_amdgcn_wmma_f32_16x16x32_bf16(
            false, ahi[mt], false, blo, (short)0, acc[mt][nt], false, false);
        acc[mt][nt] = __builtin_amdgcn_wmma_f32_16x16x32_bf16(
            false, alo[mt], false, bhi, (short)0, acc[mt][nt], false, false);
      }
    }
  }

  // ---- epilogue: C/D layout — VGPR r: M = r + 8*(lane>>4), N = lane&15 ----
  const int mOff = ((lane >> 4) << 3);
  const int nCol = lane & 15;

  if (MODE == 0) {
    float lsum = 0.0f;
#pragma unroll
    for (int mt = 0; mt < 4; ++mt)
#pragma unroll
      for (int nt = 0; nt < 2; ++nt)
#pragma unroll
        for (int r = 0; r < 8; ++r) {
          const int row = blockM + wm * 64 + mt * 16 + r + mOff;
          const int col = blockN + wn * 32 + nt * 16 + nCol;
          const float p = tanhf(acc[mt][nt][r]);
          const float e = xin[(size_t)row * FEATS + col] - p;
          lsum += e * e;
        }
    // wave32 reduction, then cross-wave via LDS, one atomic per block
#pragma unroll
    for (int off = 16; off; off >>= 1) lsum += __shfl_xor(lsum, off, 32);
    __syncthreads();                       // LDS tiles no longer needed
    float* red = (float*)sAhi;
    if (lane == 0) red[wave] = lsum;
    __syncthreads();
    if (t == 0) {
      float s = 0.0f;
#pragma unroll
      for (int i = 0; i < 8; ++i) s += red[i];
      atomicAdd(sumsq, s);
    }
  } else {
    const float nscale = noise_scale_from(sumsq);
#pragma unroll
    for (int mt = 0; mt < 4; ++mt)
#pragma unroll
      for (int nt = 0; nt < 2; ++nt)
#pragma unroll
        for (int r = 0; r < 8; ++r) {
          const int row = blockM + wm * 64 + mt * 16 + r + mOff;
          const int col = blockN + wn * 32 + nt * 16 + nCol;
          const size_t idx = (size_t)row * FEATS + col;
          const float grad = acc[mt][nt][r] - xin[idx];          // x@J - field
          const float nz = gauss_noise(((uint64_t)(step + 1) << 36) + idx);
          out[idx] = tanhf(A[idx] - grad * DT_C + nz * nscale);  // A == x_cur
        }
  }
}

// Step 0: x = 0 so x@J = 0 -> pure elementwise, skips one full GEMM.
__global__ __launch_bounds__(256)
void step0_kernel(const float* __restrict__ xin, float* __restrict__ out,
                  const float* __restrict__ sumsq)
{
  const float nscale = noise_scale_from(sumsq);
  const size_t n = (size_t)TOKENS * FEATS;
  for (size_t i = (size_t)blockIdx.x * blockDim.x + threadIdx.x; i < n;
       i += (size_t)gridDim.x * blockDim.x) {
    const float nz = gauss_noise(((uint64_t)1 << 36) + i);
    out[i] = tanhf(xin[i] * DT_C + nz * nscale);
  }
}

__global__ void init_scalar_kernel(float* s) { *s = 0.0f; }

extern "C" void kernel_launch(void* const* d_in, const int* in_sizes, int n_in,
                              void* d_out, int out_size, void* d_ws, size_t ws_size,
                              hipStream_t stream) {
  const float* x_input = (const float*)d_in[0];   // [8192, 4096] f32
  const float* W       = (const float*)d_in[1];   // [4096, 4096] f32
  const float* J       = (const float*)d_in[2];   // [4096, 4096] f32
  float* out   = (float*)d_out;                   // [8192, 4096] f32
  float* sumsq = (float*)d_ws;                    // 1 float scratch scalar
  float* xbuf  = (float*)((char*)d_ws + 256);     // 128 MB ping buffer

  const dim3 grid(FEATS / BN, TOKENS / BM);       // (32, 64)
  const dim3 block(256);

  // free_energy = mean((x - tanh(x@W))^2), accumulated into *sumsq
  init_scalar_kernel<<<1, 1, 0, stream>>>(sumsq);
  fused_gemm_kernel<0><<<grid, block, 0, stream>>>(x_input, W, x_input,
                                                   nullptr, sumsq, 0);

  // Langevin: step 0 elementwise (x starts at zero), then 9 GEMM steps
  // ping-pong between ws and d_out; step 9 (odd) lands in d_out.
  step0_kernel<<<4096, block, 0, stream>>>(x_input, xbuf, sumsq);
  float* cur = xbuf;
  for (int s = 1; s < 10; ++s) {
    float* nxt = (s & 1) ? out : xbuf;
    fused_gemm_kernel<1><<<grid, block, 0, stream>>>(cur, J, x_input,
                                                     nxt, sumsq, s);
    cur = nxt;
  }
}
